// DAN_1211180777570
// MI455X (gfx1250) — matrix operand order repaired and checked by reading, not compiled, not run on hardware
//
#include <hip/hip_runtime.h>
#include <hip/hip_bf16.h>

typedef __attribute__((ext_vector_type(16))) _Float16 v16h;
typedef __attribute__((ext_vector_type(8)))  _Float16 v8h;
typedef __attribute__((ext_vector_type(8)))  float    v8f;
typedef __attribute__((ext_vector_type(4)))  unsigned int v4u;
typedef __attribute__((ext_vector_type(8)))  int v8i;
typedef __attribute__((ext_vector_type(4)))  int v4i;

constexpr int Bp = 8, Np = 100, Tp = 32, Ep = 512, Hp = 256, Rp = 49, FVp = 512;
constexpr int BN  = Bp * Np;   // 800 tweets
constexpr int BNT = BN * Tp;   // 25600 (tweet,t) rows
constexpr int BNR = BN * Rp;   // 39200 (tweet,region) rows

// ---------------- WMMA fragment helpers (gfx1250 wave32 layouts) ----------------

__device__ __forceinline__ v16h mk16(v8h lo, v8h hi) {
  return __builtin_shufflevector(lo, hi, 0,1,2,3,4,5,6,7,8,9,10,11,12,13,14,15);
}

// A-matrix 16x32 f16: lanes 0-15 hold rows M=lane, K=0..7 / K=16..23;
// lanes 16-31 hold K=8..15 / K=24..31 of the same rows.
__device__ __forceinline__ v16h load_a(const _Float16* Arow0, int lda, int k0, int lane) {
  int m  = lane & 15;
  int ko = k0 + ((lane >> 4) << 3);
  const _Float16* p = Arow0 + (size_t)m * lda + ko;
  v8h lo = *(const v8h*)(p);
  v8h hi = *(const v8h*)(p + 16);
  return mk16(lo, hi);
}

// B-matrix 32x16 f16 for C = A @ W^T with W row-major [N,K]:
// lane n (0-15) holds column n = W row n, K=0..7 / K=16..23; lanes 16-31 hold K=8..15 / K=24..31.
__device__ __forceinline__ v16h load_b(const _Float16* W, int ldw, int n0, int k0, int lane) {
  int n  = n0 + (lane & 15);
  int ko = k0 + ((lane >> 4) << 3);
  const _Float16* p = W + (size_t)n * ldw + ko;
  v8h lo = *(const v8h*)(p);
  v8h hi = *(const v8h*)(p + 16);
  return mk16(lo, hi);
}

__device__ __forceinline__ v8f wmma16(v16h a, v16h b, v8f c) {
  // 8-arg form: (neg_a, A, neg_b, B, c_mod, C, reuse_a, reuse_b)
  return __builtin_amdgcn_wmma_f32_16x16x32_f16(false, a, false, b, (short)0, c, false, false);
}

__device__ __forceinline__ float sigmoidf_(float x) { return 1.f / (1.f + __expf(-x)); }

// LDS byte offset = low 32 bits of the generic address (ISA 10.2: LDS aperture
// maps addr[31:0] directly onto the LDS offset).
__device__ __forceinline__ unsigned lds_off(const void* p) {
  return (unsigned)(unsigned long long)p;
}

// ---------------- Tensor Data Mover: async 2D f16 tile -> LDS ----------------
// Builds D# group0/group1 per CDNA5 ISA 8.3/8.4 (count=1, type=2 "image",
// data_size=1 (2 bytes), 2D tile: tile_dim0=cols, tile_dim1=rows, groups 2/3 zero).
__device__ __forceinline__ void tdm_load_2d_f16(unsigned lds_byte_off,
                                                const _Float16* gsrc,
                                                unsigned rows, unsigned cols,
                                                unsigned row_stride_elems) {
  unsigned long long ga = (unsigned long long)gsrc;
  v4u g0;
  g0[0] = 1u;                                              // count=1, user descriptor
  g0[1] = lds_byte_off;                                    // lds_addr (bytes)
  g0[2] = (unsigned)(ga & 0xffffffffu);                    // global_addr[31:0]
  g0[3] = (unsigned)((ga >> 32) & 0x1ffffffu) | (2u << 30);// global_addr[56:32] | type=2
  v8i g1;
  g1[0] = (int)(1u << 16);                                 // data_size=1 (2B); no mask/pad
  g1[1] = (int)((cols & 0xffffu) << 16);                   // tensor_dim0[15:0]
  g1[2] = (int)(((cols >> 16) & 0xffffu) | (0xffffu << 16));// tensor_dim0 hi | tensor_dim1 lo
  g1[3] = (int)((cols & 0xffffu) << 16);                   // tensor_dim1 hi=0 | tile_dim0
  g1[4] = (int)(rows & 0xffffu);                           // tile_dim1=rows, tile_dim2=0
  g1[5] = (int)row_stride_elems;                           // tensor_dim0_stride[31:0]
  g1[6] = 0;                                               // stride hi | dim1_stride lo
  g1[7] = 0;
  v4i z4 = {0, 0, 0, 0};
#if defined(__clang_major__) && (__clang_major__ >= 23)
  v8i z8 = {0, 0, 0, 0, 0, 0, 0, 0};
  __builtin_amdgcn_tensor_load_to_lds(g0, g1, z4, z4, z8, 0);
#else
  __builtin_amdgcn_tensor_load_to_lds(g0, g1, z4, z4, 0);
#endif
}

// ---------------- elementwise / gather kernels ----------------

__global__ void cast_f16_kernel(_Float16* __restrict__ dst, const float* __restrict__ src, int n) {
  int i = blockIdx.x * blockDim.x + threadIdx.x;
  if (i < n) dst[i] = (_Float16)src[i];
}

__global__ void embed_kernel(_Float16* __restrict__ Xh, const int* __restrict__ tok,
                             const float* __restrict__ emb, int total) {
  int i = blockIdx.x * blockDim.x + threadIdx.x;
  if (i >= total) return;
  int r = i >> 9, e = i & 511;
  int id = tok[r];
  Xh[i] = (_Float16)emb[(size_t)id * Ep + e];
}

__global__ void vmean_kernel(_Float16* __restrict__ vm, const float* __restrict__ V, int total) {
  int i = blockIdx.x * blockDim.x + threadIdx.x;
  if (i >= total) return;
  int tw = i >> 9, e = i & 511;
  const float* vp = V + ((size_t)tw * Rp) * FVp + e;
  float s = 0.f;
  for (int r = 0; r < Rp; ++r) s += vp[(size_t)r * FVp];
  vm[i] = (_Float16)(s * (1.f / 49.f));
}

// m0 = mean_t(u) * v0
__global__ void m0_kernel(float* __restrict__ m0, const float* __restrict__ Uf,
                          const float* __restrict__ v0, int total) {
  int i = blockIdx.x * blockDim.x + threadIdx.x;
  if (i >= total) return;
  int tw = i >> 9, e = i & 511;
  const float* up = Uf + ((size_t)tw * Tp) * Ep + e;
  float s = 0.f;
  for (int t = 0; t < Tp; ++t) s += up[(size_t)t * Ep];
  m0[i] = (s * (1.f / 32.f)) * v0[i];
}

// ---------------- generic WMMA GEMM: C = epi(A @ W^T + bias) ----------------
template <int EPI>  // 0=none, 1=tanh
__global__ void gemm_wmma_kernel(float* __restrict__ Cf, _Float16* __restrict__ Ch,
                                 const _Float16* __restrict__ A, const _Float16* __restrict__ W,
                                 const float* __restrict__ bias,
                                 int M, int N, int K, int lda, int ldw, int ldc) {
  int lane   = threadIdx.x & 31;
  int wave   = (int)((blockIdx.x * blockDim.x + threadIdx.x) >> 5);
  int nwaves = (int)((gridDim.x * blockDim.x) >> 5);
  int ntb    = N >> 6;
  int jobs   = (M >> 4) * ntb;
  for (int job = wave; job < jobs; job += nwaves) {
    int mt = job / ntb, nb = job - mt * ntb;
    const _Float16* Arow = A + (size_t)mt * 16 * lda;
    int n0 = nb * 64;
    v8f acc0 = {}, acc1 = {}, acc2 = {}, acc3 = {};
    for (int k = 0; k < K; k += 32) {
      if (k + 64 < K) __builtin_prefetch(Arow + (size_t)(lane & 15) * lda + k + 64, 0, 0);
      v16h a = load_a(Arow, lda, k, lane);
      acc0 = wmma16(a, load_b(W, ldw, n0,      k, lane), acc0);
      acc1 = wmma16(a, load_b(W, ldw, n0 + 16, k, lane), acc1);
      acc2 = wmma16(a, load_b(W, ldw, n0 + 32, k, lane), acc2);
      acc3 = wmma16(a, load_b(W, ldw, n0 + 48, k, lane), acc3);
    }
    v8f accs[4] = {acc0, acc1, acc2, acc3};
    int nlane = lane & 15;
    int mbase = mt * 16 + ((lane >> 4) << 3);
    for (int j = 0; j < 4; ++j) {
      int n = n0 + j * 16 + nlane;
      float bb = bias ? bias[n] : 0.f;
#pragma unroll
      for (int r = 0; r < 8; ++r) {
        float v = accs[j][r] + bb;
        if (EPI == 1) v = tanhf(v);
        size_t idx = (size_t)(mbase + r) * ldc + n;
        if (Cf) Cf[idx] = v;
        if (Ch) Ch[idx] = (_Float16)v;
      }
    }
  }
}

// ---------------- BiLSTM: 16 tweets/WG, 512 threads (16 waves) ----------------
// Wave w owns N-tiles {G*16 tiles}: acc0..3 = i/f/g/o, all at hidden cols [16w,16w+16).
// x_t tiles are TDM-streamed into a double-buffered LDS stage.
__global__ void lstm_wmma_kernel(float* __restrict__ U, const _Float16* __restrict__ X,
                                 const _Float16* __restrict__ Wih, const _Float16* __restrict__ Whh,
                                 const float* __restrict__ bias, int reverse) {
  __shared__ __align__(16) _Float16 h_sh[16 * Hp];       // recurrent state (A-matrix)
  __shared__ __align__(16) _Float16 xbuf[2][16 * Ep];    // TDM double buffer for x_t
  int tid = threadIdx.x, lane = tid & 31, wv = tid >> 5; // 16 waves
  int tweet0 = blockIdx.x * 16;
  for (int i = tid; i < 16 * Hp; i += blockDim.x) h_sh[i] = (_Float16)0.f;

  int nlane = lane & 15, mlo = (lane >> 4) << 3;
  int col0  = wv * 16 + nlane;  // hidden column this wave owns
  float bi = bias[col0], bf = bias[256 + col0], bg = bias[512 + col0], bo = bias[768 + col0];
  v8f cst = {};
  const v8f vzero = {};

  int tfirst = reverse ? (Tp - 1) : 0;
  if (wv == 0)
    tdm_load_2d_f16(lds_off(&xbuf[0][0]),
                    X + ((size_t)tweet0 * Tp + tfirst) * Ep, 16, Ep, Tp * Ep);

  for (int s = 0; s < Tp; ++s) {
    int t = reverse ? (Tp - 1 - s) : s;
    int cur = s & 1;
    if (wv == 0) {
      if (s + 1 < Tp) {
        int tn = reverse ? (Tp - 2 - s) : (s + 1);
        tdm_load_2d_f16(lds_off(&xbuf[cur ^ 1][0]),
                        X + ((size_t)tweet0 * Tp + tn) * Ep, 16, Ep, Tp * Ep);
        __builtin_amdgcn_s_wait_tensorcnt(1);  // xbuf[cur] landed, next in flight
      } else {
        __builtin_amdgcn_s_wait_tensorcnt(0);
      }
    }
    __syncthreads();  // xbuf[cur] ready + prev h_sh writes visible

    v8f acc0 = vzero, acc1 = vzero, acc2 = vzero, acc3 = vzero;
    for (int k = 0; k < Ep; k += 32) {           // g += x_t @ Wih^T  (A from LDS)
      v16h a = load_a(&xbuf[cur][0], Ep, k, lane);
      acc0 = wmma16(a, load_b(Wih, Ep,       wv * 16, k, lane), acc0);
      acc1 = wmma16(a, load_b(Wih, Ep, 256 + wv * 16, k, lane), acc1);
      acc2 = wmma16(a, load_b(Wih, Ep, 512 + wv * 16, k, lane), acc2);
      acc3 = wmma16(a, load_b(Wih, Ep, 768 + wv * 16, k, lane), acc3);
    }
    for (int k = 0; k < Hp; k += 32) {           // g += h @ Whh^T
      v16h a = load_a(h_sh, Hp, k, lane);
      acc0 = wmma16(a, load_b(Whh, Hp,       wv * 16, k, lane), acc0);
      acc1 = wmma16(a, load_b(Whh, Hp, 256 + wv * 16, k, lane), acc1);
      acc2 = wmma16(a, load_b(Whh, Hp, 512 + wv * 16, k, lane), acc2);
      acc3 = wmma16(a, load_b(Whh, Hp, 768 + wv * 16, k, lane), acc3);
    }
    __syncthreads();  // all waves finished reading h_sh

    v8f hv = vzero;
#pragma unroll
    for (int r = 0; r < 8; ++r) {
      float ig = sigmoidf_(acc0[r] + bi);
      float fg = sigmoidf_(acc1[r] + bf);
      float gg = tanhf    (acc2[r] + bg);
      float og = sigmoidf_(acc3[r] + bo);
      float c  = fg * cst[r] + ig * gg;
      cst[r] = c;
      hv[r]  = og * tanhf(c);
    }
#pragma unroll
    for (int r = 0; r < 8; ++r) {
      int m = mlo + r;
      h_sh[m * Hp + col0] = (_Float16)hv[r];
      U[(((size_t)(tweet0 + m)) * Tp + t) * Ep + (reverse ? Hp : 0) + col0] = hv[r];
    }
    // loop-top barrier orders these writes against the next step's reads
  }
}

// ---------------- textual attention scan (32 steps, 16 tweets/WG, 512 threads) ----------------
__global__ void attnu_kernel(float* __restrict__ Mu, const float* __restrict__ M0,
                             const _Float16* __restrict__ Hu, const float* __restrict__ Uf,
                             const _Float16* __restrict__ Wum, const float* __restrict__ Wuh) {
  __shared__ __align__(16) float    m_f[16 * Ep];
  __shared__ __align__(16) _Float16 m_h[16 * Ep];
  __shared__ __align__(16) _Float16 h_h[16 * Ep];
  __shared__ __align__(16) _Float16 hub[2][16 * Ep];   // TDM double buffer: Hu[:,t,:]
  __shared__ float sc[16 * Tp];
  __shared__ float aw[16 * Tp];
  int tid = threadIdx.x, lane = tid & 31, wv = tid >> 5;  // 16 waves
  int tweet0 = blockIdx.x * 16;
  for (int i = tid; i < 16 * Ep; i += blockDim.x) {
    float v = M0[(size_t)tweet0 * Ep + i];
    m_f[i] = v; m_h[i] = (_Float16)v;
  }
  int nlane = lane & 15, mlo = (lane >> 4) << 3;

  if (wv == 0)
    tdm_load_2d_f16(lds_off(&hub[0][0]), Hu + ((size_t)tweet0 * Tp) * Ep, 16, Ep, Tp * Ep);

  for (int t = 0; t < Tp; ++t) {
    int cur = t & 1;
    if (wv == 0) {
      if (t + 1 < Tp) {
        tdm_load_2d_f16(lds_off(&hub[cur ^ 1][0]),
                        Hu + ((size_t)tweet0 * Tp + (t + 1)) * Ep, 16, Ep, Tp * Ep);
        __builtin_amdgcn_s_wait_tensorcnt(1);
      } else {
        __builtin_amdgcn_s_wait_tensorcnt(0);
      }
    }
    __syncthreads();  // m from prev step + hub[cur] ready
    // P1: z = m @ Wum^T (WMMA); h = tanh(z) * Hu[t]
    for (int j = 0; j < 2; ++j) {
      int n0 = (wv * 2 + j) * 16;
      v8f acc = {};
      for (int k = 0; k < Ep; k += 32)
        acc = wmma16(load_a(m_h, Ep, k, lane), load_b(Wum, Ep, n0, k, lane), acc);
      int n = n0 + nlane;
#pragma unroll
      for (int r = 0; r < 8; ++r) {
        int m = mlo + r;
        h_h[m * Ep + n] = (_Float16)(tanhf(acc[r]) * (float)hub[cur][m * Ep + n]);
      }
    }
    __syncthreads();
    // P2: scores[tw][j] = h . Wuh[j]
    for (int p = tid; p < 16 * Tp; p += blockDim.x) {
      int tw = p >> 5, jj = p & 31;
      const float*    wr = Wuh + (size_t)jj * Ep;
      const _Float16* hr = h_h + (size_t)tw * Ep;
      float s = 0.f;
      for (int e = 0; e < Ep; ++e) s += (float)hr[e] * wr[e];
      sc[p] = s;
    }
    __syncthreads();
    // P3: softmax over T
    if (tid < 16) {
      float mx = -1e30f;
      for (int jj = 0; jj < Tp; ++jj) mx = fmaxf(mx, sc[tid * Tp + jj]);
      float ss = 0.f;
      for (int jj = 0; jj < Tp; ++jj) { float e = __expf(sc[tid * Tp + jj] - mx); aw[tid * Tp + jj] = e; ss += e; }
      float inv = 1.f / ss;
      for (int jj = 0; jj < Tp; ++jj) aw[tid * Tp + jj] *= inv;
    }
    __syncthreads();
    // P4: m += a @ u1
    for (int p = tid; p < 16 * Ep; p += blockDim.x) {
      int tw = p >> 9, e = p & 511;
      const float* ur = Uf + (((size_t)(tweet0 + tw)) * Tp) * Ep + e;
      float s = 0.f;
      for (int tt = 0; tt < Tp; ++tt) s += aw[tw * Tp + tt] * ur[(size_t)tt * Ep];
      float nm = m_f[p] + s;
      m_f[p] = nm; m_h[p] = (_Float16)nm;
    }
  }
  __syncthreads();
  for (int i = tid; i < 16 * Ep; i += blockDim.x)
    Mu[(size_t)tweet0 * Ep + i] = m_f[i];
}

// ---------------- visual attention scan (49 steps, 16 tweets/WG, 512 threads) ----------------
__global__ void attnv_kernel(float* __restrict__ Mv, const float* __restrict__ M0,
                             const _Float16* __restrict__ Hv, const float* __restrict__ V,
                             const _Float16* __restrict__ Wvm, const float* __restrict__ Wvh,
                             const _Float16* __restrict__ Ph) {
  __shared__ __align__(16) float    m_f[16 * Ep];
  __shared__ __align__(16) _Float16 m_h[16 * Ep];
  __shared__ __align__(16) _Float16 h_h[16 * Ep];
  __shared__ __align__(16) _Float16 av_h[16 * FVp];
  __shared__ __align__(16) _Float16 hvb[2][16 * Ep];   // TDM double buffer: Hv[:,n,:]
  __shared__ float sc[16 * Rp];
  __shared__ float aw[16 * Rp];
  int tid = threadIdx.x, lane = tid & 31, wv = tid >> 5;  // 16 waves
  int tweet0 = blockIdx.x * 16;
  for (int i = tid; i < 16 * Ep; i += blockDim.x) {
    float v = M0[(size_t)tweet0 * Ep + i];
    m_f[i] = v; m_h[i] = (_Float16)v;
  }
  int nlane = lane & 15, mlo = (lane >> 4) << 3;

  if (wv == 0)
    tdm_load_2d_f16(lds_off(&hvb[0][0]), Hv + ((size_t)tweet0 * Rp) * Ep, 16, Ep, Rp * Ep);

  for (int nstep = 0; nstep < Rp; ++nstep) {
    int cur = nstep & 1;
    if (wv == 0) {
      if (nstep + 1 < Rp) {
        tdm_load_2d_f16(lds_off(&hvb[cur ^ 1][0]),
                        Hv + ((size_t)tweet0 * Rp + (nstep + 1)) * Ep, 16, Ep, Rp * Ep);
        __builtin_amdgcn_s_wait_tensorcnt(1);
      } else {
        __builtin_amdgcn_s_wait_tensorcnt(0);
      }
    }
    __syncthreads();
    // P1: z = m @ Wvm^T (WMMA); h = tanh(z) * Hv[n]
    for (int j = 0; j < 2; ++j) {
      int n0 = (wv * 2 + j) * 16;
      v8f acc = {};
      for (int k = 0; k < Ep; k += 32)
        acc = wmma16(load_a(m_h, Ep, k, lane), load_b(Wvm, Ep, n0, k, lane), acc);
      int n = n0 + nlane;
#pragma unroll
      for (int r = 0; r < 8; ++r) {
        int m = mlo + r;
        h_h[m * Ep + n] = (_Float16)(tanhf(acc[r]) * (float)hvb[cur][m * Ep + n]);
      }
    }
    __syncthreads();
    // P2: scores[tw][j] = h . Wvh[j]
    for (int p = tid; p < 16 * Rp; p += blockDim.x) {
      int tw = p / Rp, jj = p - tw * Rp;
      const float*    wr = Wvh + (size_t)jj * Ep;
      const _Float16* hr = h_h + (size_t)tw * Ep;
      float s = 0.f;
      for (int e = 0; e < Ep; ++e) s += (float)hr[e] * wr[e];
      sc[p] = s;
    }
    __syncthreads();
    // P3: softmax over R=49
    if (tid < 16) {
      float mx = -1e30f;
      for (int jj = 0; jj < Rp; ++jj) mx = fmaxf(mx, sc[tid * Rp + jj]);
      float ss = 0.f;
      for (int jj = 0; jj < Rp; ++jj) { float e = __expf(sc[tid * Rp + jj] - mx); aw[tid * Rp + jj] = e; ss += e; }
      float inv = 1.f / ss;
      for (int jj = 0; jj < Rp; ++jj) aw[tid * Rp + jj] *= inv;
    }
    __syncthreads();
    // P4: av = a @ v1 (f16 A-matrix for the P projection)
    for (int p = tid; p < 16 * FVp; p += blockDim.x) {
      int tw = p >> 9, e = p & 511;
      const float* vr = V + (((size_t)(tweet0 + tw)) * Rp) * FVp + e;
      float s = 0.f;
      for (int r = 0; r < Rp; ++r) s += aw[tw * Rp + r] * vr[(size_t)r * FVp];
      av_h[p] = (_Float16)s;
    }
    __syncthreads();
    // P5: m += tanh(av @ P^T) (WMMA, in-register epilogue)
    for (int j = 0; j < 2; ++j) {
      int n0 = (wv * 2 + j) * 16;
      v8f acc = {};
      for (int k = 0; k < FVp; k += 32)
        acc = wmma16(load_a(av_h, FVp, k, lane), load_b(Ph, FVp, n0, k, lane), acc);
      int n = n0 + nlane;
#pragma unroll
      for (int r = 0; r < 8; ++r) {
        int m = mlo + r;
        float nm = m_f[m * Ep + n] + tanhf(acc[r]);
        m_f[m * Ep + n] = nm;
        m_h[m * Ep + n] = (_Float16)nm;
      }
    }
  }
  __syncthreads();
  for (int i = tid; i < 16 * Ep; i += blockDim.x)
    Mv[(size_t)tweet0 * Ep + i] = m_f[i];
}

// ---------------- classifier head: one WG per user ----------------
__global__ void head_kernel(float* __restrict__ out, const float* __restrict__ Mu,
                            const float* __restrict__ Mv, const float* __restrict__ Wc1,
                            const float* __restrict__ bc1, const float* __restrict__ Wc2,
                            const float* __restrict__ bc2) {
  __shared__ float mcat[2 * Ep];
  __shared__ float hh[Ep];
  int b = blockIdx.x, tid = threadIdx.x;
  for (int j = tid; j < 2 * Ep; j += blockDim.x) {
    const float* src = (j < Ep) ? (Mu + ((size_t)b * Np) * Ep + j)
                                : (Mv + ((size_t)b * Np) * Ep + (j - Ep));
    float s = 0.f;
    for (int n = 0; n < Np; ++n) s += src[(size_t)n * Ep];
    mcat[j] = s * (1.f / Np);
  }
  __syncthreads();
  for (int i = tid; i < Ep; i += blockDim.x) {
    const float* w = Wc1 + (size_t)i * (2 * Ep);
    float s = bc1[i];
    for (int j = 0; j < 2 * Ep; ++j) s += w[j] * mcat[j];
    hh[i] = fmaxf(s, 0.f);
  }
  __syncthreads();
  if (tid < 2) {
    const float* w = Wc2 + (size_t)tid * Ep;
    float s = bc2[tid];
    for (int j = 0; j < Ep; ++j) s += w[j] * hh[j];
    out[b * 2 + tid] = s;
  }
}

// ---------------- host orchestration ----------------
extern "C" void kernel_launch(void* const* d_in, const int* in_sizes, int n_in,
                              void* d_out, int out_size, void* d_ws, size_t ws_size,
                              hipStream_t stream) {
  (void)in_sizes; (void)n_in; (void)out_size; (void)ws_size;
  const int*   tokens = (const int*)d_in[0];
  const float* images = (const float*)d_in[1];
  const float* emb    = (const float*)d_in[2];
  const float* Wih_f  = (const float*)d_in[3];
  const float* Whh_f  = (const float*)d_in[4];
  const float* b_f    = (const float*)d_in[5];
  const float* Wih_b  = (const float*)d_in[6];
  const float* Whh_b  = (const float*)d_in[7];
  const float* b_b    = (const float*)d_in[8];
  const float* Wu     = (const float*)d_in[9];
  const float* Wum    = (const float*)d_in[10];
  const float* Wuh    = (const float*)d_in[11];
  const float* Wv     = (const float*)d_in[12];
  const float* Wvm    = (const float*)d_in[13];
  const float* Wvh    = (const float*)d_in[14];
  const float* P      = (const float*)d_in[15];
  const float* Wc1    = (const float*)d_in[16];
  const float* bc1    = (const float*)d_in[17];
  const float* Wc2    = (const float*)d_in[18];
  const float* bc2    = (const float*)d_in[19];
  float* out = (float*)d_out;

  char* wsb = (char*)d_ws;
  size_t off = 0;
  auto alloc = [&](size_t bytes) -> void* {
    void* p = (void*)(wsb + off);
    off += (bytes + 255) & ~((size_t)255);
    return p;
  };

  _Float16* Wihf_h = (_Float16*)alloc((size_t)4 * Hp * Ep * 2);
  _Float16* Whhf_h = (_Float16*)alloc((size_t)4 * Hp * Hp * 2);
  _Float16* Wihb_h = (_Float16*)alloc((size_t)4 * Hp * Ep * 2);
  _Float16* Whhb_h = (_Float16*)alloc((size_t)4 * Hp * Hp * 2);
  _Float16* Wu_h   = (_Float16*)alloc((size_t)Ep * Ep * 2);
  _Float16* Wum_h  = (_Float16*)alloc((size_t)Ep * Ep * 2);
  _Float16* Wv_h   = (_Float16*)alloc((size_t)Ep * FVp * 2);
  _Float16* Wvm_h  = (_Float16*)alloc((size_t)Ep * Ep * 2);
  _Float16* P_h    = (_Float16*)alloc((size_t)Ep * FVp * 2);
  _Float16* X_h    = (_Float16*)alloc((size_t)BNT * Ep * 2);
  float*    U_f    = (float*)   alloc((size_t)BNT * Ep * 4);
  _Float16* U_h    = (_Float16*)alloc((size_t)BNT * Ep * 2);
  _Float16* Hu_h   = (_Float16*)alloc((size_t)BNT * Ep * 2);
  _Float16* V_h    = (_Float16*)alloc((size_t)BNR * FVp * 2);
  _Float16* Hv_h   = (_Float16*)alloc((size_t)BNR * Ep * 2);
  _Float16* Vm_h   = (_Float16*)alloc((size_t)BN * FVp * 2);
  float*    v0_f   = (float*)   alloc((size_t)BN * Ep * 4);
  float*    m0_f   = (float*)   alloc((size_t)BN * Ep * 4);
  float*    mu_f   = (float*)   alloc((size_t)BN * Ep * 4);
  float*    mv_f   = (float*)   alloc((size_t)BN * Ep * 4);

  const int thr = 256;
  auto blocks = [](int n, int t) { return (n + t - 1) / t; };

  // 1) weights -> f16
  cast_f16_kernel<<<blocks(4*Hp*Ep, thr), thr, 0, stream>>>(Wihf_h, Wih_f, 4*Hp*Ep);
  cast_f16_kernel<<<blocks(4*Hp*Hp, thr), thr, 0, stream>>>(Whhf_h, Whh_f, 4*Hp*Hp);
  cast_f16_kernel<<<blocks(4*Hp*Ep, thr), thr, 0, stream>>>(Wihb_h, Wih_b, 4*Hp*Ep);
  cast_f16_kernel<<<blocks(4*Hp*Hp, thr), thr, 0, stream>>>(Whhb_h, Whh_b, 4*Hp*Hp);
  cast_f16_kernel<<<blocks(Ep*Ep,  thr), thr, 0, stream>>>(Wu_h,  Wu,  Ep*Ep);
  cast_f16_kernel<<<blocks(Ep*Ep,  thr), thr, 0, stream>>>(Wum_h, Wum, Ep*Ep);
  cast_f16_kernel<<<blocks(Ep*FVp, thr), thr, 0, stream>>>(Wv_h,  Wv,  Ep*FVp);
  cast_f16_kernel<<<blocks(Ep*Ep,  thr), thr, 0, stream>>>(Wvm_h, Wvm, Ep*Ep);
  cast_f16_kernel<<<blocks(Ep*FVp, thr), thr, 0, stream>>>(P_h,   P,   Ep*FVp);

  // 2) embedding gather -> f16, images -> f16
  embed_kernel<<<blocks(BNT*Ep, thr), thr, 0, stream>>>(X_h, tokens, emb, BNT*Ep);
  cast_f16_kernel<<<blocks(BNR*FVp, thr), thr, 0, stream>>>(V_h, images, BNR*FVp);

  // 3) BiLSTM (forward + backward): 512 threads, TDM-streamed x_t
  lstm_wmma_kernel<<<BN/16, 512, 0, stream>>>(U_f, X_h, Wihf_h, Whhf_h, b_f, 0);
  lstm_wmma_kernel<<<BN/16, 512, 0, stream>>>(U_f, X_h, Wihb_h, Whhb_h, b_b, 1);
  cast_f16_kernel<<<blocks(BNT*Ep, thr), thr, 0, stream>>>(U_h, U_f, BNT*Ep);

  // 4) big projections: Hu = tanh(U@Wu^T), Hv = tanh(V@Wv^T)
  gemm_wmma_kernel<1><<<1600, thr, 0, stream>>>(nullptr, Hu_h, U_h, Wu_h, nullptr,
                                                BNT, Ep, Ep, Ep, Ep, Ep);
  gemm_wmma_kernel<1><<<2450, thr, 0, stream>>>(nullptr, Hv_h, V_h, Wv_h, nullptr,
                                                BNR, Ep, FVp, FVp, FVp, Ep);

  // 5) m0 = mean_t(u) * tanh(mean_r(v) @ P^T)
  vmean_kernel<<<blocks(BN*FVp, thr), thr, 0, stream>>>(Vm_h, images, BN*FVp);
  gemm_wmma_kernel<1><<<50, thr, 0, stream>>>(v0_f, nullptr, Vm_h, P_h, nullptr,
                                              BN, Ep, FVp, FVp, FVp, Ep);
  m0_kernel<<<blocks(BN*Ep, thr), thr, 0, stream>>>(m0_f, U_f, v0_f, BN*Ep);

  // 6) dual attention scans with TDM-streamed Hu/Hv tiles
  attnu_kernel<<<BN/16, 512, 0, stream>>>(mu_f, m0_f, Hu_h, U_f, Wum_h, Wuh);
  attnv_kernel<<<BN/16, 512, 0, stream>>>(mv_f, m0_f, Hv_h, images, Wvm_h, Wvh, P_h);

  // 7) per-user mean + MLP head
  head_kernel<<<Bp, thr, 0, stream>>>(out, mu_f, mv_f, Wc1, bc1, Wc2, bc2);
}
